// lstm_cell_level_16131897164393
// MI455X (gfx1250) — compile-verified, matrix operand order checked by reading, not run on hardware
//
#include <hip/hip_runtime.h>
#include <hip/hip_bf16.h>

// ---------------------------------------------------------------------------
// LSTM (B=1024, T=128, I=H=1024, C=10) for gfx1250 (MI455X), wave32 + WMMA.
//   gates_t = [x_t | h] @ [W_ih ; W_hh]^T + (b_ih + b_hh)      (K = 2048)
//   Weights packed bf16, K-major per output column, 4 gates interleaved per
//   hidden unit (col = 4*u + gate) -> LSTM cell fused in GEMM epilogue.
//   Staging: GLOBAL_LOAD_ASYNC_TO_LDS_B128 + ASYNCcnt, 3-stage LDS pipeline
//   with static stage offsets and a single barrier per K-chunk.
// ---------------------------------------------------------------------------

typedef __attribute__((ext_vector_type(16))) __bf16 v16bf;
typedef __attribute__((ext_vector_type(4)))  __bf16 v4bf;
typedef __attribute__((ext_vector_type(8)))  float  v8f;

union FragAB {
    v16bf v;
    uint4 q[2];
};

#define LDA 40              // padded LDS row pitch in bf16 elems (80 B, conflict-free)
#define STAGE_BYTES 20480u  // A tile (10240 B) + B tile (10240 B)

__device__ __forceinline__ float fsig(float x) {
    return __builtin_amdgcn_rcpf(1.0f + __builtin_amdgcn_exp2f(-1.44269504088896f * x));
}
__device__ __forceinline__ float ftanh(float x) {
    return 2.0f * fsig(2.0f * x) - 1.0f;
}

// gfx1250 async global->LDS copy, 16 bytes per lane. VDST holds the per-lane
// LDS byte address; low 32 bits of a generic LDS pointer are exactly that.
__device__ __forceinline__ void async_b128(unsigned lds_off, const void* gptr) {
    asm volatile("global_load_async_to_lds_b128 %0, %1, off"
                 :: "v"(lds_off), "v"(gptr)
                 : "memory");
}

// ------------------------------ pack kernels -------------------------------

__global__ void pack_x_kernel(const float* __restrict__ x, __bf16* __restrict__ xb) {
    size_t i = (size_t)blockIdx.x * blockDim.x + threadIdx.x;   // one float4 per thread
    float4 v = ((const float4*)x)[i];
    v4bf o = { (__bf16)v.x, (__bf16)v.y, (__bf16)v.z, (__bf16)v.w };
    ((v4bf*)xb)[i] = o;
}

// Wp[n][k], n = 4*u + gate (gate order i,f,g,o), k in [0,2048): k<1024 -> W_ih else W_hh.
__global__ void pack_w_kernel(const float* __restrict__ Wih, const float* __restrict__ Whh,
                              __bf16* __restrict__ Wp) {
    int idx = blockIdx.x * 256 + threadIdx.x;    // 4096*2048 total
    int k = idx & 2047;
    int n = idx >> 11;
    int g = n & 3;
    int u = n >> 2;
    float v = (k < 1024) ? Wih[(size_t)(g * 1024 + u) * 1024 + k]
                         : Whh[(size_t)(g * 1024 + u) * 1024 + (k - 1024)];
    Wp[idx] = (__bf16)v;
}

__global__ void pack_state_kernel(const float* __restrict__ h0, const float* __restrict__ c0,
                                  const float* __restrict__ bih, const float* __restrict__ bhh,
                                  __bf16* __restrict__ hb, float* __restrict__ c,
                                  float* __restrict__ bias) {
    int i = blockIdx.x * 256 + threadIdx.x;      // 1048576 threads
    hb[i] = (__bf16)h0[i];
    c[i] = c0[i];
    if (i < 4096) {
        int g = i & 3, u = i >> 2;
        bias[i] = bih[g * 1024 + u] + bhh[g * 1024 + u];
    }
}

// ------------------------------ LSTM step ----------------------------------
// grid = (32, 8): blockIdx.x -> 128 packed gate columns, blockIdx.y -> 128 batch rows.
// block = 256 threads = 8 waves, 2 (M) x 4 (N); wave tile 64x32 = 4x2 WMMA tiles.

__global__ __launch_bounds__(256)
void lstm_step_kernel(const __bf16* __restrict__ X, const __bf16* __restrict__ Wp,
                      const float* __restrict__ bias, const __bf16* __restrict__ h_in,
                      __bf16* __restrict__ h_out, float* __restrict__ c, int t) {
    __shared__ char smem[65536];   // 3 stages x 20480 B in K loop; 64 KB f32 tile in epilogue

    const int tid   = threadIdx.x;
    const int lane  = tid & 31;
    const int wave  = tid >> 5;
    const int waveM = wave >> 2;   // 0..1
    const int waveN = wave & 3;    // 0..3
    const int m0    = blockIdx.y * 128;
    const int n0    = blockIdx.x * 128;
    const int half  = lane >> 4;
    const int l16   = lane & 15;

    // --- staging decomposition: 512 16B chunks per tile, 2 per thread per tile
    const int cA = tid, cB = tid + 256;
    const int r0 = cA >> 2, s0 = cA & 3;
    const int r1 = cB >> 2, s1 = cB & 3;

    // global base pointers (kc added per iteration)
    const __bf16* aX0 = X + ((size_t)((m0 + r0) * 128 + t)) * 1024 + s0 * 8;
    const __bf16* aX1 = X + ((size_t)((m0 + r1) * 128 + t)) * 1024 + s1 * 8;
    const __bf16* aH0 = h_in + (size_t)(m0 + r0) * 1024 + s0 * 8;
    const __bf16* aH1 = h_in + (size_t)(m0 + r1) * 1024 + s1 * 8;
    const __bf16* bW0 = Wp + (size_t)(n0 + r0) * 2048 + s0 * 8;
    const __bf16* bW1 = Wp + (size_t)(n0 + r1) * 2048 + s1 * 8;

    // per-lane LDS byte addresses for the async copies (stage 0)
    const unsigned ldsBase = (unsigned)(uintptr_t)(void*)smem;
    const unsigned ldsA0 = ldsBase + r0 * 80 + s0 * 16;
    const unsigned ldsA1 = ldsBase + r1 * 80 + s1 * 16;
    const unsigned ldsB0 = ldsBase + 10240 + r0 * 80 + s0 * 16;
    const unsigned ldsB1 = ldsBase + 10240 + r1 * 80 + s1 * 16;

    v8f zero = {0.f, 0.f, 0.f, 0.f, 0.f, 0.f, 0.f, 0.f};
    v8f acc[4][2];
#pragma unroll
    for (int a = 0; a < 4; ++a)
#pragma unroll
        for (int b = 0; b < 2; ++b) acc[a][b] = zero;

    // ---- fragment math + 8 WMMAs from one LDS stage
    auto compute_from = [&](unsigned compOff) {
        const char* stg = smem + compOff;
        const __bf16* AshS = (const __bf16*)stg;
        const __bf16* BshS = (const __bf16*)(stg + 10240);
        FragAB afrag[4];
#pragma unroll
        for (int a = 0; a < 4; ++a) {
            int row = waveM * 64 + a * 16 + l16;
            const uint4* p = (const uint4*)(AshS + row * LDA + half * 8);
            afrag[a].q[0] = p[0];
            afrag[a].q[1] = p[2];            // +32 bytes
        }
        FragAB bfrag[2];
#pragma unroll
        for (int b = 0; b < 2; ++b) {
            int col = waveN * 32 + b * 16 + l16;
            const uint4* p = (const uint4*)(BshS + col * LDA + half * 8);
            bfrag[b].q[0] = p[0];
            bfrag[b].q[1] = p[2];
        }
#pragma unroll
        for (int a = 0; a < 4; ++a)
#pragma unroll
            for (int b = 0; b < 2; ++b)
                acc[a][b] = __builtin_amdgcn_wmma_f32_16x16x32_bf16(
                    false, afrag[a].v, false, bfrag[b].v,
                    (short)0, acc[a][b], false, false);
    };

    // ---- one pipeline step: issue chunk kc+32 into issOff, compute chunk kc from compOff
    auto do_step = [&](int kc, unsigned compOff, unsigned issOff) {
        const int kn = kc + 32;
        const __bf16* a0 = (kn < 1024) ? (aX0 + kn) : (aH0 + (kn - 1024));
        const __bf16* a1 = (kn < 1024) ? (aX1 + kn) : (aH1 + (kn - 1024));
        async_b128(ldsA0 + issOff, a0);
        async_b128(ldsA1 + issOff, a1);
        async_b128(ldsB0 + issOff, bW0 + kn);
        async_b128(ldsB1 + issOff, bW1 + kn);
        asm volatile("s_wait_asynccnt 4" ::: "memory");   // chunk-kc group complete
        __syncthreads();                                   // stage compOff visible to all waves
        compute_from(compOff);
    };

    // ---- prologue: stage K-chunk 0 into stage 0
    async_b128(ldsA0, aX0);
    async_b128(ldsA1, aX1);
    async_b128(ldsB0, bW0);
    async_b128(ldsB1, bW1);

    // ---- 63 pipelined chunks as 21 supersteps of 3 (static stage offsets)
    int kc = 0;
#pragma unroll 1
    for (int s3 = 0; s3 < 21; ++s3) {
        do_step(kc,      0u,              STAGE_BYTES);
        do_step(kc + 32, STAGE_BYTES,     2u * STAGE_BYTES);
        do_step(kc + 64, 2u * STAGE_BYTES, 0u);
        kc += 96;
    }
    // ---- tail: chunk 63 (stage 0), nothing left to issue
    asm volatile("s_wait_asynccnt 0" ::: "memory");
    __syncthreads();
    compute_from(0u);
    __syncthreads();   // all fragment reads done before tile overwrites stages

    // ---- epilogue: dump f32 tile to LDS, then fused LSTM cell
    float* tile = (float*)smem;   // [128][128] f32 = 64 KB
#pragma unroll
    for (int a = 0; a < 4; ++a)
#pragma unroll
        for (int b = 0; b < 2; ++b) {
            int mBase = waveM * 64 + a * 16 + half * 8;
            int n = waveN * 32 + b * 16 + l16;
#pragma unroll
            for (int r = 0; r < 8; ++r)
                tile[(mBase + r) * 128 + n] = acc[a][b][r];
        }
    __syncthreads();

#pragma unroll
    for (int s = 0; s < 16; ++s) {
        int idx = tid + s * 256;     // 0..4095 = 128 rows x 32 hidden units
        int row = idx >> 5;
        int j   = idx & 31;
        float4 g4 = *(float4*)(tile + row * 128 + j * 4);
        float4 b4 = *(const float4*)(bias + n0 + j * 4);
        float ig = fsig(g4.x + b4.x);
        float fg = fsig(g4.y + b4.y);
        float gg = ftanh(g4.z + b4.z);
        float og = fsig(g4.w + b4.w);
        int u  = (n0 >> 2) + j;
        int bg = m0 + row;
        size_t ci = (size_t)bg * 1024 + u;
        float cn = fg * c[ci] + ig * gg;
        c[ci] = cn;
        h_out[ci] = (__bf16)(og * ftanh(cn));
    }
}

// --------------------------- final projection ------------------------------

__global__ __launch_bounds__(256)
void proj_kernel(const __bf16* __restrict__ h, const float* __restrict__ Wout,
                 const float* __restrict__ bout, float* __restrict__ out) {
    __shared__ float red[256];
    int b = blockIdx.x;
    int tid = threadIdx.x;
    float acc[10];
#pragma unroll
    for (int cc = 0; cc < 10; ++cc) acc[cc] = 0.f;
#pragma unroll
    for (int s = 0; s < 4; ++s) {
        int u = tid + s * 256;
        float hv = (float)h[(size_t)b * 1024 + u];
#pragma unroll
        for (int cc = 0; cc < 10; ++cc) acc[cc] += hv * Wout[cc * 1024 + u];
    }
    for (int cc = 0; cc < 10; ++cc) {
        red[tid] = acc[cc];
        __syncthreads();
        for (int ofs = 128; ofs > 0; ofs >>= 1) {
            if (tid < ofs) red[tid] += red[tid + ofs];
            __syncthreads();
        }
        if (tid == 0) out[b * 10 + cc] = red[0] + bout[cc];
        __syncthreads();
    }
}

// ------------------------------ launcher -----------------------------------

extern "C" void kernel_launch(void* const* d_in, const int* in_sizes, int n_in,
                              void* d_out, int out_size, void* d_ws, size_t ws_size,
                              hipStream_t stream) {
    const float* x    = (const float*)d_in[0];   // [B,T,I]
    const float* h0   = (const float*)d_in[1];   // [B,H]
    const float* c0   = (const float*)d_in[2];   // [B,H]
    const float* Wih  = (const float*)d_in[3];   // [4H,I]
    const float* bih  = (const float*)d_in[4];   // [4H]
    const float* Whh  = (const float*)d_in[5];   // [4H,H]
    const float* bhh  = (const float*)d_in[6];   // [4H]
    const float* Wout = (const float*)d_in[7];   // [C,H]
    const float* bout = (const float*)d_in[8];   // [C]

    char* ws = (char*)d_ws;
    __bf16* Xb   = (__bf16*)(ws);                      // 268,435,456 B
    __bf16* Wp   = (__bf16*)(ws + 268435456);          //  16,777,216 B
    float*  bias = (float*) (ws + 285212672);          //      16,384 B
    __bf16* hA   = (__bf16*)(ws + 285229056);          //   2,097,152 B
    __bf16* hB   = (__bf16*)(ws + 287326208);          //   2,097,152 B
    float*  cbuf = (float*) (ws + 289423360);          //   4,194,304 B

    pack_x_kernel<<<131072, 256, 0, stream>>>(x, Xb);          // 1024*128*1024/4 float4s
    pack_w_kernel<<<32768, 256, 0, stream>>>(Wih, Whh, Wp);    // 4096*2048 elems
    pack_state_kernel<<<4096, 256, 0, stream>>>(h0, c0, bih, bhh, hA, cbuf, bias);

    dim3 grid(32, 8);
    for (int t = 0; t < 128; ++t) {
        const __bf16* hin  = (t & 1) ? hB : hA;
        __bf16*       hout = (t & 1) ? hA : hB;
        lstm_step_kernel<<<grid, 256, 0, stream>>>(Xb, Wp, bias, hin, hout, cbuf, t);
    }
    // T=128 even -> final hidden state is in hA
    proj_kernel<<<1024, 256, 0, stream>>>(hA, Wout, bout, (float*)d_out);
}
